// GCN_69191923138833
// MI455X (gfx1250) — compile-verified, matrix operand order checked by reading, not run on hardware
//
#include <hip/hip_runtime.h>

// ---------------- problem constants (match reference) ----------------
#define Nn     100000      // nodes
#define Ee     400000      // directed edges
#define ETOT   500000      // edges + self loops
#define Gg     4096        // graphs
#define EMBD   300
#define EMBP   320         // padded (multiple of 32 for WMMA K, 16 for N)
#define FEATD  256
#define NLAYER 5

// ---------------- WMMA / TDM types ----------------
typedef __attribute__((ext_vector_type(16))) __bf16        v16bf;
typedef __attribute__((ext_vector_type(8)))  float         v8f;
typedef __attribute__((ext_vector_type(4)))  int           vi4;
typedef __attribute__((ext_vector_type(4)))  unsigned int  u32x4;
typedef __attribute__((ext_vector_type(8)))  int           i32x8;
typedef __attribute__((ext_vector_type(4)))  int           i32x4;

union Frag { vi4 q[2]; v16bf v; };   // 16 bf16 = 8 dwords per lane

#if defined(__has_builtin)
#if __has_builtin(__builtin_amdgcn_tensor_load_to_lds) && \
    __has_builtin(__builtin_amdgcn_s_wait_tensorcnt)
#define HAVE_TDM 1
#endif
#endif

__device__ __forceinline__ unsigned short f2bf(float x) {
  unsigned int u = __float_as_uint(x);
  u += 0x7fffu + ((u >> 16) & 1u);          // round-to-nearest-even
  return (unsigned short)(u >> 16);
}
__device__ __forceinline__ unsigned int pack2(float a, float b) {
  return (unsigned int)f2bf(a) | ((unsigned int)f2bf(b) << 16);
}

// ---------------- initial node embedding: h = emb1[x0] + emb2[x1] (bf16, padded) ----
__global__ void embed_kernel(const int* __restrict__ xn,
                             const float* __restrict__ e1,
                             const float* __restrict__ e2,
                             unsigned short* __restrict__ hb) {
  int i = blockIdx.x, c = threadIdx.x;     // block = node, 320 threads
  float v = 0.f;
  if (c < EMBD) {
    int a = xn[i * 2], b = xn[i * 2 + 1];
    v = e1[(size_t)a * EMBD + c] + e2[(size_t)b * EMBD + c];
  }
  hb[(size_t)i * EMBP + c] = f2bf(v);
}

// ---------------- pre-swizzle B into WMMA fragment order -------------------
// out layout: [Nt][Kt][lane(32)][vreg(8)] dwords, each dword = bf16 pair (k, k+1)
// ISA 16-bit layout: lane L -> col n = L%16; vreg v -> k = kt*32+(v/4)*16+(L/16)*8+(v%4)*2
__global__ void prep_wfrag_kernel(const float* __restrict__ W,
                                  unsigned int* __restrict__ F,
                                  int Ksrc, int Nsrc, int Kt, int Nt) {
  int idx = blockIdx.x * blockDim.x + threadIdx.x;
  int total = Nt * Kt * 32 * 8;
  if (idx >= total) return;
  int v    = idx & 7;
  int lane = (idx >> 3) & 31;
  int rest = idx >> 8;
  int kt   = rest % Kt;
  int nt   = rest / Kt;
  int n = nt * 16 + (lane & 15);
  int k = kt * 32 + (v >> 2) * 16 + (lane >> 4) * 8 + (v & 3) * 2;
  float w0 = (k     < Ksrc && n < Nsrc) ? W[(size_t)k       * Nsrc + n] : 0.f;
  float w1 = (k + 1 < Ksrc && n < Nsrc) ? W[(size_t)(k + 1) * Nsrc + n] : 0.f;
  F[idx] = pack2(w0, w1);
}

// ---------------- WMMA GEMM: C[M,ldc] = A[M,lda](bf16) @ Bfrag -------------
// Block = 8 waves = 128 rows. Per n-group the 4*KT KB B panel is DMA'd into
// LDS by the Tensor Data Mover (wave 0 issues tensor_load_to_lds, waits on
// TENSORcnt, workgroup barrier publishes). A fragments live in registers.
// Inner loop is pure ds_load_b128 + v_wmma. NT % 4 == 0 (compile-time).
template <int KT, int NT>
__global__ __launch_bounds__(256) void wmma_gemm_kernel(
    const unsigned short* __restrict__ A, const unsigned int* __restrict__ B,
    float* __restrict__ C, int Mtiles, int lda, int ldc) {
  __shared__ unsigned int sB[KT * 4 * 256];            // <= 40 KB

  const int wave = threadIdx.x >> 5;
  const int lane = threadIdx.x & 31;
  const int rowTile = blockIdx.x * 8 + wave;
  const bool active = rowTile < Mtiles;                // wave-uniform
  const int half = lane >> 4;
  const int mrow = (rowTile << 4) + (lane & 15);

  // preload all A fragments for this wave (16 rows x KT k-tiles)
  Frag afr[KT];
  if (active) {
    const unsigned int* Arow =
        reinterpret_cast<const unsigned int*>(A + (size_t)mrow * lda);
    #pragma unroll
    for (int kt = 0; kt < KT; ++kt) {
      const vi4* pa = reinterpret_cast<const vi4*>(Arow + (kt << 4) + (half << 2));
      afr[kt].q[0] = pa[0];        // dwords [base .. base+3]
      afr[kt].q[1] = pa[2];        // dwords [base+8 .. base+11]
    }
  }

  #pragma unroll
  for (int ng = 0; ng < NT / 4; ++ng) {
    // ---- stage 4 n-tiles of B into LDS ----
    __syncthreads();                                   // previous group consumed
#ifdef HAVE_TDM
    if (wave == 0) {
      // 1-D D# descriptor: KT*512 contiguous 8-byte elements, global -> LDS
      const unsigned nelem = (unsigned)(KT * 512);
      unsigned long long ga =
          (unsigned long long)(const char*)(B + (size_t)ng * 4 * KT * 256);
      unsigned ldsoff = (unsigned)(size_t)(&sB[0]);    // low 32b of LDS aperture addr
      u32x4 g0 = { 1u,                                 // count=1, user descriptor
                   ldsoff,                             // lds_addr (bytes)
                   (unsigned)(ga & 0xffffffffu),       // global_addr[31:0]
                   (unsigned)((ga >> 32) & 0x01ffffffu) | (2u << 30) }; // [56:32] | type=2
      i32x8 g1 = { (int)(3u << 16),                    // wg_mask=0, data_size=3 (8B)
                   (int)((nelem & 0xffffu) << 16),     // tensor_dim0[15:0]
                   (int)((nelem >> 16) | (1u << 16)),  // tensor_dim0[31:16] | tensor_dim1=1
                   (int)((nelem & 0xffffu) << 16),     // tile_dim0
                   0,                                  // tile_dim1=0 (1-D), tile_dim2=0
                   (int)nelem,                         // tensor_dim0_stride[31:0]
                   0, 0 };
      i32x4 gz = { 0, 0, 0, 0 };
#if __clang_major__ >= 23
      i32x8 gz8 = { 0, 0, 0, 0, 0, 0, 0, 0 };
      __builtin_amdgcn_tensor_load_to_lds(g0, g1, gz, gz, gz8, 0);
#else
      __builtin_amdgcn_tensor_load_to_lds(g0, g1, gz, gz, 0);
#endif
      __builtin_amdgcn_s_wait_tensorcnt(0);            // DMA complete
    }
#else
    {
      const vi4* src = reinterpret_cast<const vi4*>(B) + (size_t)ng * 4 * KT * 64;
      vi4* dst = reinterpret_cast<vi4*>(sB);
      #pragma unroll
      for (int i = 0; i < KT; ++i)
        dst[threadIdx.x + i * 256] = src[threadIdx.x + i * 256];
    }
#endif
    __syncthreads();

    if (active) {
      v8f c0 = {}, c1 = {}, c2 = {}, c3 = {};
      #pragma unroll
      for (int kt = 0; kt < KT; ++kt) {
        Frag b0, b1, b2, b3;
        const vi4* base = reinterpret_cast<const vi4*>(sB) + (lane << 1);
        const vi4* p0 = base + (0 * KT + kt) * 64;
        const vi4* p1 = base + (1 * KT + kt) * 64;
        const vi4* p2 = base + (2 * KT + kt) * 64;
        const vi4* p3 = base + (3 * KT + kt) * 64;
        b0.q[0] = p0[0]; b0.q[1] = p0[1];
        b1.q[0] = p1[0]; b1.q[1] = p1[1];
        b2.q[0] = p2[0]; b2.q[1] = p2[1];
        b3.q[0] = p3[0]; b3.q[1] = p3[1];
        c0 = __builtin_amdgcn_wmma_f32_16x16x32_bf16(
            false, afr[kt].v, false, b0.v, (short)0, c0, false, false);
        c1 = __builtin_amdgcn_wmma_f32_16x16x32_bf16(
            false, afr[kt].v, false, b1.v, (short)0, c1, false, false);
        c2 = __builtin_amdgcn_wmma_f32_16x16x32_bf16(
            false, afr[kt].v, false, b2.v, (short)0, c2, false, false);
        c3 = __builtin_amdgcn_wmma_f32_16x16x32_bf16(
            false, afr[kt].v, false, b3.v, (short)0, c3, false, false);
      }
      // ---- store 16x64 strip; D layout: vgpr r -> row r + half*8, col = lane%16
      const int rowBase = (rowTile << 4) + (half << 3);
      const int colBase = (ng << 6) + (lane & 15);
      #pragma unroll
      for (int r = 0; r < 8; ++r) {
        float* crow = C + (size_t)(rowBase + r) * ldc + colBase;
        crow[0]  = c0[r];
        crow[16] = c1[r];
        crow[32] = c2[r];
        crow[48] = c3[r];
      }
    }
  }
}

// ---------------- edge message + segment-sum (scatter-add) -----------------
// block = 320 threads (channel-parallel), walks 32 edges; fully coalesced.
// 10 lanes prefetch the next edge's source row (one per 128B cacheline).
__global__ void scatter_kernel(const float* __restrict__ hw, float* __restrict__ acc,
                               const int* __restrict__ ei, const int* __restrict__ ea,
                               const float* __restrict__ e1, const float* __restrict__ e2,
                               int l) {
  const int c = threadIdx.x;
  const int base = blockIdx.x * 32;
  for (int t = 0; t < 32; ++t) {
    int j = base + t;
    if (j >= ETOT) return;                              // block-uniform
    int row, col, a0, a1;
    if (j < Ee) {
      row = ei[j]; col = ei[Ee + j];
      a0 = ea[j * 2]; a1 = ea[j * 2 + 1];
    } else {                                            // appended self-loop
      row = j - Ee; col = row; a0 = 4; a1 = 0;
    }
    // prefetch next edge's gather row (global_prefetch)
    int jn = j + 1;
    if (jn < ETOT && (c & 31) == 0 && c < EMBD) {
      int rown = (jn < Ee) ? ei[jn] : (jn - Ee);
      __builtin_prefetch(&hw[(size_t)rown * EMBP + c], 0, 3);
    }
    float e = e1[l * 5 + a0] + e2[l * 3 + a1];          // scalar edge embedding
    if (c < EMBD) {
      float msg = hw[(size_t)row * EMBP + c] + e;
      atomicAdd(&acc[(size_t)col * EMBP + c], msg);
    }
  }
}

// ---------------- BatchNorm: pass 1 (per-channel sum / sumsq) --------------
__global__ void bn_stats_kernel(const float* __restrict__ src, float* __restrict__ stats,
                                int M, int C, int ld, int rowsPerBlock) {
  int c = threadIdx.x;
  if (c >= C) return;
  int r0 = blockIdx.x * rowsPerBlock;
  int r1 = min(M, r0 + rowsPerBlock);
  float s = 0.f, ss = 0.f;
  for (int r = r0; r < r1; ++r) {
    float x = src[(size_t)r * ld + c];
    s += x; ss += x * x;
  }
  atomicAdd(&stats[c], s);
  atomicAdd(&stats[C + c], ss);
}

// ---------------- BatchNorm: pass 2 (normalize [+relu], write bf16 [+f32]) --
__global__ void bn_apply_kernel(const float* __restrict__ src, const float* __restrict__ stats,
                                const float* __restrict__ gamma, const float* __restrict__ beta,
                                unsigned short* __restrict__ dst_bf, float* __restrict__ dst_f32,
                                int C, int ld, int do_relu, float invM) {
  int r = blockIdx.x, c = threadIdx.x;                  // block = row, blockDim = ld
  float y = 0.f;
  if (c < C) {
    float mean = stats[c] * invM;
    float var  = stats[C + c] * invM - mean * mean;     // biased variance
    float x = src[(size_t)r * ld + c];
    y = (x - mean) * rsqrtf(var + 1e-5f) * gamma[c] + beta[c];
    if (do_relu) y = fmaxf(y, 0.f);
  }
  dst_bf[(size_t)r * ld + c] = f2bf(y);
  if (dst_f32) dst_f32[(size_t)r * ld + c] = y;
}

// ---------------- global mean pool -----------------------------------------
__global__ void pool_kernel(const float* __restrict__ hf, const int* __restrict__ batch,
                            float* __restrict__ pool, float* __restrict__ cnt) {
  const int c = threadIdx.x;                            // 320 threads
  const int base = blockIdx.x * 32;
  for (int t = 0; t < 32; ++t) {
    int i = base + t;
    if (i >= Nn) return;                                // block-uniform
    int g = batch[i];
    if (c < EMBD) atomicAdd(&pool[(size_t)g * EMBD + c], hf[(size_t)i * EMBP + c]);
    if (c == EMBD) atomicAdd(&cnt[g], 1.0f);
  }
}

__global__ void pool_div_kernel(float* __restrict__ pool, const float* __restrict__ cnt) {
  int g = blockIdx.x, c = threadIdx.x;                  // blockDim = 300
  pool[(size_t)g * EMBD + c] *= 1.0f / fmaxf(cnt[g], 1.0f);
}

// ---------------- f32 -> bf16 (pad + optional bias) -------------------------
__global__ void bf16_pad_kernel(const float* __restrict__ src, unsigned short* __restrict__ dst,
                                int Csrc, int ldsrc, const float* __restrict__ bias) {
  int r = blockIdx.x, c = threadIdx.x;                  // blockDim = padded width
  float v = 0.f;
  if (c < Csrc) {
    v = src[(size_t)r * ldsrc + c];
    if (bias) v += bias[c];
  }
  dst[(size_t)r * blockDim.x + c] = f2bf(v);
}

__global__ void bias_add_kernel(float* __restrict__ dst, const float* __restrict__ b) {
  dst[(size_t)blockIdx.x * blockDim.x + threadIdx.x] += b[threadIdx.x];
}

// ===========================================================================
extern "C" void kernel_launch(void* const* d_in, const int* in_sizes, int n_in,
                              void* d_out, int out_size, void* d_ws, size_t ws_size,
                              hipStream_t stream) {
  const int*   x_nodes    = (const int*)  d_in[0];
  const int*   edge_index = (const int*)  d_in[1];
  const int*   edge_attr  = (const int*)  d_in[2];
  const int*   batch      = (const int*)  d_in[3];
  const float* x_emb1     = (const float*)d_in[4];
  const float* x_emb2     = (const float*)d_in[5];
  const float* gcn_W      = (const float*)d_in[6];
  // gcn_b (d_in[7]) cancels exactly through BN mean-subtraction
  const float* edge_emb1  = (const float*)d_in[8];
  const float* edge_emb2  = (const float*)d_in[9];
  const float* bn_gamma   = (const float*)d_in[10];
  const float* bn_beta    = (const float*)d_in[11];
  const float* feat_W     = (const float*)d_in[12];
  const float* feat_b     = (const float*)d_in[13];
  const float* out_W1     = (const float*)d_in[14];
  // out_b1 (15), out_b2 (17) cancel through their BNs
  const float* out_W2     = (const float*)d_in[16];
  const float* out_W3     = (const float*)d_in[18];
  const float* out_b3     = (const float*)d_in[19];
  const float* bn1_g      = (const float*)d_in[20];
  const float* bn1_b      = (const float*)d_in[21];
  const float* bn2_g      = (const float*)d_in[22];
  const float* bn2_b      = (const float*)d_in[23];

  float* h_pool  = (float*)d_out;                         // [G, 300]
  float* out_mlp = (float*)d_out + (size_t)Gg * EMBD;     // [G, 256]

  // ---- workspace layout (offsets 256B-aligned) ----
  char* ws = (char*)d_ws;
  unsigned short* hb     = (unsigned short*)(ws);               // [N,320] bf16  64.0 MB
  float*          hw     = (float*)(ws + 64000000);             // [N,320] f32  128.0 MB
  float*          acc    = (float*)(ws + 192000000);            // [N,320] f32  128.0 MB
  unsigned int*   wfrag  = (unsigned int*)(ws + 320000000);     // <=204800 B
  float*          stats  = (float*)(ws + 320204800);            // 640 f32
  float*          cnt    = (float*)(ws + 320207360);            // 4096 f32
  unsigned short* hpoolp = (unsigned short*)(ws + 320223744);   // [G,320] bf16
  unsigned short* habuf  = (unsigned short*)(ws + 322845184);   // [G,256] bf16
  float*          tbuf   = (float*)(ws + 324942336);            // [G,256] f32

  // ---- initial embedding ----
  embed_kernel<<<Nn, EMBP, 0, stream>>>(x_nodes, x_emb1, x_emb2, hb);

  // ---- 5 GCN layers ----
  for (int l = 0; l < NLAYER; ++l) {
    prep_wfrag_kernel<<<20 * 10, 256, 0, stream>>>(
        gcn_W + (size_t)l * EMBD * EMBD, wfrag, EMBD, EMBD, 10, 20);
    wmma_gemm_kernel<10, 20><<<(6250 + 7) / 8, 256, 0, stream>>>(
        hb, wfrag, hw, 6250, EMBP, EMBP);
    hipMemsetAsync(acc, 0, (size_t)Nn * EMBP * sizeof(float), stream);
    scatter_kernel<<<(ETOT + 31) / 32, EMBP, 0, stream>>>(
        hw, acc, edge_index, edge_attr, edge_emb1, edge_emb2, l);
    hipMemsetAsync(stats, 0, 640 * sizeof(float), stream);
    bn_stats_kernel<<<(Nn + 127) / 128, EMBP, 0, stream>>>(acc, stats, Nn, EMBD, EMBP, 128);
    bn_apply_kernel<<<Nn, EMBP, 0, stream>>>(
        acc, stats, bn_gamma + l * EMBD, bn_beta + l * EMBD,
        hb, (l == NLAYER - 1) ? hw : nullptr,        // last layer: keep f32 h for pooling
        EMBD, EMBP, (l < NLAYER - 1) ? 1 : 0, 1.0f / Nn);
  }

  // ---- global mean pool (h_pool is output #1) ----
  hipMemsetAsync(h_pool, 0, (size_t)Gg * EMBD * sizeof(float), stream);
  hipMemsetAsync(cnt, 0, Gg * sizeof(float), stream);
  pool_kernel<<<(Nn + 31) / 32, EMBP, 0, stream>>>(hw, batch, h_pool, cnt);
  pool_div_kernel<<<Gg, EMBD, 0, stream>>>(h_pool, cnt);

  // ---- MLP head ----
  bf16_pad_kernel<<<Gg, EMBP, 0, stream>>>(h_pool, hpoolp, EMBD, EMBD, nullptr);
  prep_wfrag_kernel<<<16 * 10, 256, 0, stream>>>(feat_W, wfrag, EMBD, FEATD, 10, 16);
  wmma_gemm_kernel<10, 16><<<32, 256, 0, stream>>>(hpoolp, wfrag, tbuf, Gg / 16, EMBP, FEATD);
  bf16_pad_kernel<<<Gg, FEATD, 0, stream>>>(tbuf, habuf, FEATD, FEATD, feat_b);

  prep_wfrag_kernel<<<16 * 8, 256, 0, stream>>>(out_W1, wfrag, FEATD, FEATD, 8, 16);
  wmma_gemm_kernel<8, 16><<<32, 256, 0, stream>>>(habuf, wfrag, tbuf, Gg / 16, FEATD, FEATD);
  hipMemsetAsync(stats, 0, 640 * sizeof(float), stream);
  bn_stats_kernel<<<Gg / 64, FEATD, 0, stream>>>(tbuf, stats, Gg, FEATD, FEATD, 64);
  bn_apply_kernel<<<Gg, FEATD, 0, stream>>>(tbuf, stats, bn1_g, bn1_b, habuf, nullptr,
                                            FEATD, FEATD, 1, 1.0f / Gg);

  prep_wfrag_kernel<<<16 * 8, 256, 0, stream>>>(out_W2, wfrag, FEATD, FEATD, 8, 16);
  wmma_gemm_kernel<8, 16><<<32, 256, 0, stream>>>(habuf, wfrag, tbuf, Gg / 16, FEATD, FEATD);
  hipMemsetAsync(stats, 0, 640 * sizeof(float), stream);
  bn_stats_kernel<<<Gg / 64, FEATD, 0, stream>>>(tbuf, stats, Gg, FEATD, FEATD, 64);
  bn_apply_kernel<<<Gg, FEATD, 0, stream>>>(tbuf, stats, bn2_g, bn2_b, habuf, nullptr,
                                            FEATD, FEATD, 1, 1.0f / Gg);

  prep_wfrag_kernel<<<16 * 8, 256, 0, stream>>>(out_W3, wfrag, FEATD, FEATD, 8, 16);
  wmma_gemm_kernel<8, 16><<<32, 256, 0, stream>>>(habuf, wfrag, out_mlp, Gg / 16, FEATD, FEATD);
  bias_add_kernel<<<Gg, FEATD, 0, stream>>>(out_mlp, out_b3);
}
